// GraphVAE_r_74526272520488
// MI455X (gfx1250) — compile-verified, matrix operand order checked by reading, not run on hardware
//
#include <hip/hip_runtime.h>
#include <hip/hip_bf16.h>
#include <stdint.h>

// ---------------------------------------------------------------------------
// GraphVAE forward for MI455X (gfx1250, wave32).
//  - Dense GEMMs: v_wmma_f32_16x16x32_f16, f16 operands staged to LDS with
//    GLOBAL_LOAD_ASYNC_TO_LDS_B128, triple-buffered and software-pipelined
//    with non-zero s_wait_asynccnt thresholds (ASYNCcnt retires in order).
//  - Inputs pre-converted to zero-padded f16 buffers -> branch-free GEMM.
//  - Edge softmax/aggregation: 3 linear passes over E+N edges with
//    L2-resident gathers and float atomics (192MB L2 >> per-layer arrays).
// ---------------------------------------------------------------------------

#define D_IN 256
#define HDIM 96
#define ZDIM 32
#define TPB  256

typedef __attribute__((ext_vector_type(16))) _Float16 v16h;
typedef __attribute__((ext_vector_type(8)))  float    v8f;

union HFrag { v16h v; unsigned u[8]; _Float16 h[16]; };

// ---------------- async global->LDS copy (CDNA5), with fallback ------------
#if defined(__has_builtin)
#  if __has_builtin(__builtin_amdgcn_global_load_async_to_lds_b128)
#    define USE_ASYNC_LDS 1
#  endif
#  if __has_builtin(__builtin_amdgcn_s_wait_asynccnt)
#    define HAVE_WAIT_ASYNC_BUILTIN 1
#  endif
#endif

// builtin expects: (addrspace(1) int __vector(4)*, addrspace(3) ..., Ii, Ii)
typedef int v4i __attribute__((vector_size(16)));
typedef __attribute__((address_space(1))) v4i as1_v4i;
typedef __attribute__((address_space(3))) v4i as3_v4i;

__device__ __forceinline__ void async_copy_b128(const void* g, void* l) {
#ifdef USE_ASYNC_LDS
  __builtin_amdgcn_global_load_async_to_lds_b128((as1_v4i*)g, (as3_v4i*)l, 0, 0);
#else
  *(float4*)l = *(const float4*)g;   // synchronous fallback
#endif
}
template <int CNT>
__device__ __forceinline__ void wait_async_le() {
#ifdef USE_ASYNC_LDS
#  ifdef HAVE_WAIT_ASYNC_BUILTIN
  __builtin_amdgcn_s_wait_asynccnt(CNT);
#  else
  asm volatile("s_wait_asynccnt %0" :: "i"(CNT) : "memory");
#  endif
#endif
}

// ------------------------------- GEMM (WMMA) -------------------------------
// Y[Npad,F] = Xh[Npad,K] @ WT[F,K]^T.  Block tile 64x32, K-chunks of 32.
// 8 waves arranged 4(M) x 2(N); each wave owns one 16x16 C tile.
// Xh is zero-padded to Npad rows (multiple of 64) -> no load/store guards.
// Triple-buffered LDS: stage(it+1) issued before the tile-it wait; waves 0-3
// issue 2 async ops/tile (A+B), waves 4-7 issue 1 -> wait thresholds 2/1
// keep tile it+1's DMA in flight behind tile it's WMMA.
#define BM  64
#define BNT 32
#define BK  32
#define LDA 40   // halves; row stride 80B (16B aligned, conflict-free mod 64)
#define LDB 40

__global__ __launch_bounds__(TPB)
void k_gemm_wmma(const _Float16* __restrict__ Xh,  // [Npad, K]
                 const _Float16* __restrict__ WT,  // [F, K] (W transposed)
                 float* __restrict__ Y,            // [Npad, F]
                 int K, int F) {
  __shared__ _Float16 sA[3][BM * LDA];
  __shared__ _Float16 sB[3][BNT * LDB];
  const int tid  = threadIdx.x;
  const int lane = tid & 31;
  const int wave = tid >> 5;
  const int wm   = wave & 3;
  const int wn   = wave >> 2;
  const int bm   = blockIdx.x * BM;
  const int bn   = blockIdx.y * BNT;
  const int hlf  = lane >> 4;
  const int mrow = lane & 15;

  const int trow = tid >> 2;          // 0..63
  const int tseg = (tid & 3) * 8;     // 0,8,16,24 halves = 16B segments
  const int nk   = K / BK;

  // stage one 64x32 A tile + 32x32 B tile (f16) via b128 async copies
  auto stage = [&](int it, int buf) {
    const int k0 = it * BK;
    async_copy_b128(Xh + (int64_t)(bm + trow) * K + k0 + tseg,
                    &sA[buf][trow * LDA + tseg]);
    if (wave < 4) {                   // 32 columns x 4 segments, waves 0-3
      async_copy_b128(WT + (int64_t)(bn + trow) * K + k0 + tseg,
                      &sB[buf][trow * LDB + tseg]);
    }
  };

  stage(0, 0);
  v8f acc = {};
  for (int it = 0; it < nk; ++it) {
    if (it + 1 < nk) {
      stage(it + 1, (it + 1) % 3);    // issue next tile's DMA first...
      if (wave < 4) wait_async_le<2>();  // ...then wait only for tile `it`
      else          wait_async_le<1>();  // (ASYNCcnt retires in order)
    } else {
      wait_async_le<0>();             // last tile: drain
    }
    __syncthreads();                  // tile `it` visible to all waves
    const _Float16* A = sA[it % 3];
    const _Float16* B = sB[it % 3];

    // CDNA5 16-bit A 16x32 layout: VGPR v holds K pair
    //   kk = ((v&4)<<2) | (half<<3) | ((v&3)<<1), row M = lane&15.
    // B stored column-major (Wt) -> same contiguous-pair read per column.
    HFrag a, b;
    #pragma unroll
    for (int vv = 0; vv < 8; ++vv) {
      int kk = ((vv & 4) << 2) | (hlf << 3) | ((vv & 3) << 1);
      a.u[vv] = *(const unsigned*)&A[(wm * 16 + mrow) * LDA + kk];
      b.u[vv] = *(const unsigned*)&B[(wn * 16 + mrow) * LDB + kk];
    }
    acc = __builtin_amdgcn_wmma_f32_16x16x32_f16(
        /*neg_a=*/false, a.v, /*neg_b=*/false, b.v,
        /*c_mod=*/(short)0, acc, /*reuse_a=*/false, /*reuse_b=*/false);
  }

  // C/D layout: VGPR r -> M = r + 8*half, N = lane&15
  #pragma unroll
  for (int r = 0; r < 8; ++r) {
    int grow = bm + wm * 16 + hlf * 8 + r;
    Y[(int64_t)grow * F + bn + wn * 16 + mrow] = acc[r];
  }
}

// --------------------------- f16 convert passes ----------------------------
__global__ void k_cvt_x(const float* __restrict__ X, _Float16* __restrict__ Xh,
                        int64_t n_valid, int64_t n_total) {
  int64_t t = (int64_t)blockIdx.x * blockDim.x + threadIdx.x;
  if (t >= n_total) return;
  Xh[t] = (t < n_valid) ? (_Float16)X[t] : (_Float16)0.f;  // zero row padding
}
__global__ void k_cvt_wT(const float* __restrict__ W, _Float16* __restrict__ WT,
                         int K, int F) {
  int t = blockIdx.x * blockDim.x + threadIdx.x;
  if (t >= K * F) return;
  int k = t / F, f = t - k * F;
  WT[f * K + k] = (_Float16)W[t];
}

// --------------------------- attention scalars -----------------------------
__global__ __launch_bounds__(TPB)
void k_attn_scores(const float* __restrict__ H, const float* __restrict__ a0,
                   const float* __restrict__ a1, float* __restrict__ s0,
                   float* __restrict__ s1, int Nn, int F) {
  int warp = (blockIdx.x * TPB + threadIdx.x) >> 5;
  int lane = threadIdx.x & 31;
  if (warp >= Nn) return;
  const float* row = H + (int64_t)warp * F;
  float acc0 = 0.f, acc1 = 0.f;
  for (int f = lane; f < F; f += 32) {
    float h = row[f];
    acc0 += h * a0[f];
    acc1 += h * a1[f];
  }
  for (int off = 16; off; off >>= 1) {
    acc0 += __shfl_down(acc0, off, 32);
    acc1 += __shfl_down(acc1, off, 32);
  }
  if (lane == 0) { s0[warp] = acc0; s1[warp] = acc1; }
}

// ------------------------------ small utils --------------------------------
__global__ void k_fill_f32(float* p, float v, int64_t n) {
  int64_t t = (int64_t)blockIdx.x * blockDim.x + threadIdx.x;
  if (t < n) p[t] = v;
}
__global__ void k_fill_u32(unsigned* p, unsigned v, int64_t n) {
  int64_t t = (int64_t)blockIdx.x * blockDim.x + threadIdx.x;
  if (t < n) p[t] = v;
}
__global__ void k_init_bias(float* out, const float* __restrict__ bias,
                            int64_t total, int F) {
  int64_t t = (int64_t)blockIdx.x * blockDim.x + threadIdx.x;
  if (t < total) out[t] = bias[(int)(t % F)];
}

__device__ __forceinline__ unsigned f2ord(float f) {
  unsigned u = __float_as_uint(f);
  return (u & 0x80000000u) ? ~u : (u | 0x80000000u);
}
__device__ __forceinline__ float ord2f(unsigned o) {
  return __uint_as_float((o & 0x80000000u) ? (o & 0x7FFFFFFFu) : ~o);
}

// ------------------------------ edge passes --------------------------------
__global__ void k_edge_logits(const int* __restrict__ src, const int* __restrict__ dst,
                              int E_, int Ntot, const float* __restrict__ s0,
                              const float* __restrict__ s1, float* __restrict__ ebuf,
                              unsigned* __restrict__ mord) {
  int t = blockIdx.x * blockDim.x + threadIdx.x;
  if (t >= Ntot) return;
  int s = (t < E_) ? src[t] : (t - E_);   // self-loops appended
  int d = (t < E_) ? dst[t] : (t - E_);
  float e = s0[s] + s1[d];
  e = (e > 0.f) ? e : 0.2f * e;           // leaky_relu, slope 0.2
  ebuf[t] = e;
  atomicMax(&mord[d], f2ord(e));          // segment_max via ordered-uint
}

__global__ void k_edge_exp(const int* __restrict__ dst, int E_, int Ntot,
                           const unsigned* __restrict__ mord,
                           float* __restrict__ ebuf, float* __restrict__ ssum) {
  int t = blockIdx.x * blockDim.x + threadIdx.x;
  if (t >= Ntot) return;
  int d = (t < E_) ? dst[t] : (t - E_);
  float ex = __expf(ebuf[t] - ord2f(mord[d]));
  ebuf[t] = ex;
  atomicAdd(&ssum[d], ex);
}

__global__ void k_edge_aggr(const int* __restrict__ src, const int* __restrict__ dst,
                            int E_, int Ntot, const float* __restrict__ ebuf,
                            const float* __restrict__ ssum, const float* __restrict__ H,
                            float* __restrict__ out, int F) {
  int C = F >> 2;
  int64_t g = (int64_t)blockIdx.x * blockDim.x + threadIdx.x;
  int64_t tot = (int64_t)Ntot * C;
  if (g >= tot) return;
  int e = (int)(g / C);
  int c = (int)(g % C) * 4;
  int s = (e < E_) ? src[e] : (e - E_);
  int d = (e < E_) ? dst[e] : (e - E_);
  float alpha = ebuf[e] / ssum[d];
  const float4 hv = *(const float4*)(H + (int64_t)s * F + c);
  float* o = out + (int64_t)d * F + c;
  atomicAdd(o + 0, alpha * hv.x);
  atomicAdd(o + 1, alpha * hv.y);
  atomicAdd(o + 2, alpha * hv.z);
  atomicAdd(o + 3, alpha * hv.w);
}

// ------------------------------ batch norm ---------------------------------
__global__ __launch_bounds__(TPB)
void k_bn_stats(const float* __restrict__ X, float* __restrict__ mean,
                float* __restrict__ var, int Nn, int F) {
  int f = blockIdx.x;
  float s = 0.f, s2 = 0.f;
  for (int n = threadIdx.x; n < Nn; n += TPB) {
    float v = X[(int64_t)n * F + f];
    s += v; s2 += v * v;
  }
  __shared__ float rs[TPB], rq[TPB];
  rs[threadIdx.x] = s; rq[threadIdx.x] = s2;
  __syncthreads();
  for (int o = TPB / 2; o; o >>= 1) {
    if (threadIdx.x < o) {
      rs[threadIdx.x] += rs[threadIdx.x + o];
      rq[threadIdx.x] += rq[threadIdx.x + o];
    }
    __syncthreads();
  }
  if (threadIdx.x == 0) {
    float mu = rs[0] / (float)Nn;
    mean[f] = mu;
    var[f] = rq[0] / (float)Nn - mu * mu;   // biased, like jnp.var
  }
}

__global__ void k_bn_relu(const float* __restrict__ X, const float* __restrict__ mean,
                          const float* __restrict__ var, const float* __restrict__ g,
                          const float* __restrict__ b, const float* __restrict__ res,
                          float* __restrict__ Y, int64_t total, int F) {
  int64_t t = (int64_t)blockIdx.x * blockDim.x + threadIdx.x;
  if (t >= total) return;
  int f = (int)(t % F);
  float y = g[f] * (X[t] - mean[f]) * rsqrtf(var[f] + 1e-5f) + b[f];
  y = fmaxf(y, 0.f);
  if (res) y += res[t];
  Y[t] = y;
}

// ------------------------------ VAE pieces ---------------------------------
__global__ void k_rsample(const float* __restrict__ qm, const float* __restrict__ qs,
                          const float* __restrict__ eps, float* __restrict__ qz,
                          int64_t total) {
  int64_t t = (int64_t)blockIdx.x * blockDim.x + threadIdx.x;
  if (t >= total) return;
  float x = qs[t];
  float sp = (x > 20.f) ? x : log1pf(__expf(x));   // softplus
  qz[t] = qm[t] + (sp + 1e-6f) * eps[t];
}

__global__ __launch_bounds__(TPB)
void k_softmax_rows(const float* __restrict__ X, float* __restrict__ Y,
                    int Nn, int F) {
  int warp = (blockIdx.x * TPB + threadIdx.x) >> 5;
  int lane = threadIdx.x & 31;
  if (warp >= Nn) return;
  const float* row = X + (int64_t)warp * F;
  float* o = Y + (int64_t)warp * F;
  float m = -3.4e38f;
  for (int f = lane; f < F; f += 32) m = fmaxf(m, row[f]);
  for (int off = 16; off; off >>= 1) m = fmaxf(m, __shfl_xor(m, off, 32));
  float s = 0.f;
  for (int f = lane; f < F; f += 32) s += __expf(row[f] - m);
  for (int off = 16; off; off >>= 1) s += __shfl_xor(s, off, 32);
  float inv = 1.f / s;
  for (int f = lane; f < F; f += 32) o[f] = __expf(row[f] - m) * inv;
}

// ------------------------------ host driver --------------------------------
namespace {

struct GatBufs {
  float *lin, *s0, *s1, *ssum, *ebuf;
  unsigned* mord;
  _Float16 *Xh, *WT;
  const int *src, *dst;
  int N, Npad, E, E2;
};

static inline int gdiv(int64_t a, int b) { return (int)((a + b - 1) / b); }

static void run_gat(const float* X, int K, int F, const float* W, const float* a,
                    const float* bias, float* dest, const GatBufs& g,
                    hipStream_t stream) {
  const int N = g.N, Npad = g.Npad, E2 = g.E2;
  k_cvt_x<<<gdiv((int64_t)Npad * K, TPB), TPB, 0, stream>>>(
      X, g.Xh, (int64_t)N * K, (int64_t)Npad * K);
  k_cvt_wT<<<gdiv(K * F, TPB), TPB, 0, stream>>>(W, g.WT, K, F);
  dim3 gg((unsigned)(Npad / BM), (unsigned)(F / BNT));
  k_gemm_wmma<<<gg, TPB, 0, stream>>>(g.Xh, g.WT, g.lin, K, F);
  k_attn_scores<<<gdiv((int64_t)N * 32, TPB), TPB, 0, stream>>>(g.lin, a, a + F,
                                                                g.s0, g.s1, N, F);
  k_fill_u32<<<gdiv(N, TPB), TPB, 0, stream>>>(g.mord, 0u, N);
  k_fill_f32<<<gdiv(N, TPB), TPB, 0, stream>>>(g.ssum, 0.f, N);
  k_init_bias<<<gdiv((int64_t)N * F, TPB), TPB, 0, stream>>>(dest, bias,
                                                             (int64_t)N * F, F);
  k_edge_logits<<<gdiv(E2, TPB), TPB, 0, stream>>>(g.src, g.dst, g.E, E2,
                                                   g.s0, g.s1, g.ebuf, g.mord);
  k_edge_exp<<<gdiv(E2, TPB), TPB, 0, stream>>>(g.dst, g.E, E2, g.mord,
                                                g.ebuf, g.ssum);
  k_edge_aggr<<<gdiv((int64_t)E2 * (F >> 2), TPB), TPB, 0, stream>>>(
      g.src, g.dst, g.E, E2, g.ebuf, g.ssum, g.lin, dest, F);
}

}  // namespace

extern "C" void kernel_launch(void* const* d_in, const int* in_sizes, int n_in,
                              void* d_out, int out_size, void* d_ws, size_t ws_size,
                              hipStream_t stream) {
  const int N = in_sizes[0] / D_IN;
  const int E = in_sizes[1] / 2;
  const int E2 = E + N;
  const int Npad = (N + BM - 1) & ~(BM - 1);

  const float* x   = (const float*)d_in[0];
  const int*   ei  = (const int*)d_in[1];
  const float* eps = (const float*)d_in[2];
  const float* P[21];
  for (int i = 0; i < 21; ++i) P[i] = (const float*)d_in[3 + i];
  const float* enc1_W = P[0],  *enc1_a = P[1],  *enc1_b = P[2];
  const float* enc2_W = P[3],  *enc2_a = P[4],  *enc2_b = P[5];
  const float* mu_W   = P[6],  *mu_a   = P[7],  *mu_b   = P[8];
  const float* lv_W   = P[9],  *lv_a   = P[10], *lv_b   = P[11];
  const float* dec1_W = P[12], *dec1_a = P[13], *dec1_b = P[14];
  const float* dec2_W = P[15], *dec2_a = P[16], *dec2_b = P[17];
  const float* out_W  = P[18], *out_a  = P[19], *out_b  = P[20];
  const float* bne1_g = (const float*)d_in[24];
  const float* bne1_b = (const float*)d_in[25];
  const float* bne2_g = (const float*)d_in[26];
  const float* bne2_b = (const float*)d_in[27];
  const float* bnd1_g = (const float*)d_in[28];
  const float* bnd1_b = (const float*)d_in[29];
  const float* bnd2_g = (const float*)d_in[30];
  const float* bnd2_b = (const float*)d_in[31];

  // output regions: (q_z, q_m, q_s, recon) concatenated
  float* qz    = (float*)d_out;
  float* qm    = qz + (int64_t)N * ZDIM;
  float* qs    = qm + (int64_t)N * ZDIM;
  float* recon = qs + (int64_t)N * ZDIM;

  // workspace carve-up (element counts all multiples of 4 -> 16B aligned)
  float* w = (float*)d_ws;
  float* lin  = w;  w += (int64_t)Npad * D_IN;   // gemm output h = x@W
  float* gbuf = w;  w += (int64_t)N * D_IN;      // GAT aggregate output
  float* hA   = w;  w += (int64_t)N * HDIM;      // h0 / d0
  float* hB   = w;  w += (int64_t)N * HDIM;      // h  / d
  float* s0   = w;  w += N;
  float* s1   = w;  w += N;
  float* ssum = w;  w += N;
  unsigned* mord = (unsigned*)w;  w += N;
  float* ebuf = w;  w += E2;
  float* bmean = w; w += 256;
  float* bvar  = w; w += 256;
  _Float16* Xh = (_Float16*)w;  w += ((int64_t)Npad * D_IN) / 2;  // f16 X
  _Float16* WT = (_Float16*)w;  w += (D_IN * HDIM) / 2;           // f16 W^T
  (void)ws_size; (void)n_in; (void)out_size;

  GatBufs g{lin, s0, s1, ssum, ebuf, mord, Xh, WT, ei, ei + E, N, Npad, E, E2};
  const int64_t NH = (int64_t)N * HDIM;
  const int64_t NZ = (int64_t)N * ZDIM;

  // ---- encoder ----
  run_gat(x, D_IN, HDIM, enc1_W, enc1_a, enc1_b, gbuf, g, stream);
  k_bn_stats<<<HDIM, TPB, 0, stream>>>(gbuf, bmean, bvar, N, HDIM);
  k_bn_relu<<<gdiv(NH, TPB), TPB, 0, stream>>>(gbuf, bmean, bvar, bne1_g, bne1_b,
                                               nullptr, hA, NH, HDIM);  // h0
  run_gat(hA, HDIM, HDIM, enc2_W, enc2_a, enc2_b, gbuf, g, stream);
  k_bn_stats<<<HDIM, TPB, 0, stream>>>(gbuf, bmean, bvar, N, HDIM);
  k_bn_relu<<<gdiv(NH, TPB), TPB, 0, stream>>>(gbuf, bmean, bvar, bne2_g, bne2_b,
                                               hA, hB, NH, HDIM);       // h = h1+h0
  run_gat(hB, HDIM, ZDIM, mu_W, mu_a, mu_b, qm, g, stream);             // q_m
  run_gat(hB, HDIM, ZDIM, lv_W, lv_a, lv_b, qs, g, stream);             // q_s
  k_rsample<<<gdiv(NZ, TPB), TPB, 0, stream>>>(qm, qs, eps, qz, NZ);    // q_z

  // ---- decoder ----
  run_gat(qz, ZDIM, HDIM, dec1_W, dec1_a, dec1_b, gbuf, g, stream);
  k_bn_stats<<<HDIM, TPB, 0, stream>>>(gbuf, bmean, bvar, N, HDIM);
  k_bn_relu<<<gdiv(NH, TPB), TPB, 0, stream>>>(gbuf, bmean, bvar, bnd1_g, bnd1_b,
                                               nullptr, hA, NH, HDIM);  // d0
  run_gat(hA, HDIM, HDIM, dec2_W, dec2_a, dec2_b, gbuf, g, stream);
  k_bn_stats<<<HDIM, TPB, 0, stream>>>(gbuf, bmean, bvar, N, HDIM);
  k_bn_relu<<<gdiv(NH, TPB), TPB, 0, stream>>>(gbuf, bmean, bvar, bnd2_g, bnd2_b,
                                               hA, hB, NH, HDIM);       // d = d1+d0
  run_gat(hB, HDIM, D_IN, out_W, out_a, out_b, gbuf, g, stream);
  k_softmax_rows<<<gdiv((int64_t)N * 32, TPB), TPB, 0, stream>>>(gbuf, recon,
                                                                 N, D_IN);
}
